// Head_5093831213196
// MI455X (gfx1250) — matrix operand order repaired and checked by reading, not compiled, not run on hardware
//
#include <hip/hip_runtime.h>
#include <hip/hip_bf16.h>

typedef _Float16 v16h __attribute__((ext_vector_type(16)));
typedef _Float16 v8h  __attribute__((ext_vector_type(8)));
typedef float    v8f  __attribute__((ext_vector_type(8)));

#define SEQ 2048
#define DM  1024
#define DI  64
#define NB  8

// workspace layout, in units of _Float16 elements
#define WQF 0u
#define WKF 65536u                       // 32 kc * 4 nt * 32 lanes * 16 halves
#define WVF 131072u
#define W0F 196608u                      // 2 kc * 4 * 32 * 16 = 4096
#define QOFF 200704u
#define KOFF (200704u + 1048576u)
#define VOFF (200704u + 2097152u)        // V stored transposed: [b][64][SEQ]

union HV16 { v16h v; v8h h8[2]; _Float16 e[16]; };

__device__ __forceinline__ v8f zero8() {
    v8f z;
    #pragma unroll
    for (int i = 0; i < 8; ++i) z[i] = 0.0f;
    return z;
}

// ---------------------------------------------------------------------------
// Repack f32 weights [K x 64] into f16 WMMA B-fragments.
// B 32x16 f16 layout: lane L (n = L&15, h = L>>4) holds column n,
// element e -> K = e + 16*h  (16 contiguous halves per lane = 32B).
// frag addr: ((kc*4 + nt)*32 + lane)*16
// ---------------------------------------------------------------------------
__global__ __launch_bounds__(32) void prep_weights(const float* __restrict__ Wk,
                                                   const float* __restrict__ Wq,
                                                   const float* __restrict__ Wv,
                                                   const float* __restrict__ W0,
                                                   _Float16* __restrict__ ws16) {
    const int lane = threadIdx.x;
    const int n = lane & 15, h = lane >> 4;
    const int nt = blockIdx.y;
    const int idx = blockIdx.x;

    const float* W; _Float16* dst; int kc;
    if (idx < 32)       { W = Wq; dst = ws16 + WQF; kc = idx;      }
    else if (idx < 64)  { W = Wk; dst = ws16 + WKF; kc = idx - 32; }
    else if (idx < 96)  { W = Wv; dst = ws16 + WVF; kc = idx - 64; }
    else                { W = W0; dst = ws16 + W0F; kc = idx - 96; }

    HV16 hv;
    #pragma unroll
    for (int e = 0; e < 16; ++e)
        hv.e[e] = (_Float16)W[(size_t)(kc * 32 + e + 16 * h) * DI + nt * 16 + n];
    *(v16h*)(dst + ((size_t)(kc * 4 + nt) * 32 + lane) * 16) = hv.v;
}

// ---------------------------------------------------------------------------
// QKV projection: one 16-row tile per wave.  A 16x32 f16 layout used here:
// lane (m = L&15, h = L>>4) holds row m; element e -> K = e+16h
// => one contiguous group of 16 floats in the input row.
// ---------------------------------------------------------------------------
__global__ __launch_bounds__(32) void qkv_proj(const float* __restrict__ x,
                                               const float* __restrict__ bk,
                                               const float* __restrict__ bq,
                                               const float* __restrict__ bv,
                                               _Float16* __restrict__ ws16) {
    const int lane = threadIdx.x;
    const int m = lane & 15, h = lane >> 4, n = m;
    const int t0 = blockIdx.x * 16;
    const int b  = blockIdx.y;

    const _Float16* wf[3] = { ws16 + WQF, ws16 + WKF, ws16 + WVF };

    v8f acc[3][4];
    #pragma unroll
    for (int w = 0; w < 3; ++w)
        #pragma unroll
        for (int nt = 0; nt < 4; ++nt) acc[w][nt] = zero8();

    const float* arow = x + ((size_t)(b * SEQ + t0 + m)) * DM;

    for (int kc = 0; kc < 32; ++kc) {
        const float* ap = arow + kc * 32;
        const float4 f0 = *(const float4*)(ap + 16 * h);
        const float4 f1 = *(const float4*)(ap + 16 * h + 4);
        const float4 f2 = *(const float4*)(ap + 16 * h + 8);
        const float4 f3 = *(const float4*)(ap + 16 * h + 12);
        HV16 a;
        a.e[0]=(_Float16)f0.x; a.e[1]=(_Float16)f0.y; a.e[2]=(_Float16)f0.z; a.e[3]=(_Float16)f0.w;
        a.e[4]=(_Float16)f1.x; a.e[5]=(_Float16)f1.y; a.e[6]=(_Float16)f1.z; a.e[7]=(_Float16)f1.w;
        a.e[8]=(_Float16)f2.x; a.e[9]=(_Float16)f2.y; a.e[10]=(_Float16)f2.z; a.e[11]=(_Float16)f2.w;
        a.e[12]=(_Float16)f3.x; a.e[13]=(_Float16)f3.y; a.e[14]=(_Float16)f3.z; a.e[15]=(_Float16)f3.w;

        #pragma unroll
        for (int w = 0; w < 3; ++w) {
            #pragma unroll
            for (int nt = 0; nt < 4; ++nt) {
                HV16 bf;
                bf.v = *(const v16h*)(wf[w] + ((size_t)(kc * 4 + nt) * 32 + lane) * 16);
                acc[w][nt] = __builtin_amdgcn_wmma_f32_16x16x32_f16(
                    false, a.v, false, bf.v, (short)0, acc[w][nt], false, false);
            }
        }
    }

    // D layout: lane column n, VGPR r -> row m = r + 8h
    #pragma unroll
    for (int nt = 0; nt < 4; ++nt) {
        const int col = nt * 16 + n;
        const float bqv = bq[col], bkv = bk[col], bvv = bv[col];
        #pragma unroll
        for (int r = 0; r < 8; ++r) {
            const int t = t0 + r + 8 * h;
            ws16[QOFF + ((size_t)(b * SEQ + t)) * DI + col] =
                (_Float16)((acc[0][nt][r] + bqv) * 0.125f);     // scale = d_k^-0.5
            ws16[KOFF + ((size_t)(b * SEQ + t)) * DI + col] =
                (_Float16)(acc[1][nt][r] + bkv);
            ws16[VOFF + ((size_t)(b * DI + col)) * SEQ + t] =    // V transposed
                (_Float16)(acc[2][nt][r] + bvv);
        }
    }
}

// ---------------------------------------------------------------------------
// Flash attention + output projection: one 16-query tile per wave.
// ---------------------------------------------------------------------------
__global__ __launch_bounds__(32) void attn_kernel(const _Float16* __restrict__ ws16,
                                                  const float* __restrict__ b0,
                                                  float* __restrict__ out) {
    __shared__ _Float16 lds[16 * 64];

    const int lane = threadIdx.x;
    const int m = lane & 15, h = lane >> 4, n = m;
    const int t0 = blockIdx.x * 16;
    const int b  = blockIdx.y;

    // Q A-fragments, two 32-wide K-chunks of the 64-dim head
    HV16 Qa[2];
    #pragma unroll
    for (int c = 0; c < 2; ++c) {
        const _Float16* qp = ws16 + QOFF + ((size_t)(b * SEQ + t0 + m)) * DI + c * 32;
        Qa[c].h8[0] = *(const v8h*)(qp + 16 * h);
        Qa[c].h8[1] = *(const v8h*)(qp + 16 * h + 8);
    }

    v8f O[4];
    #pragma unroll
    for (int nt = 0; nt < 4; ++nt) O[nt] = zero8();
    float mst[8], lsum[8];
    #pragma unroll
    for (int r = 0; r < 8; ++r) { mst[r] = -3.0e38f; lsum[r] = 0.0f; }

    const int njt = (t0 + 15) / 32 + 1;     // causal 32-key tiles
    for (int jt = 0; jt < njt; ++jt) {
        const int tj = jt * 32;

        v8f S[2]; S[0] = zero8(); S[1] = zero8();
        #pragma unroll
        for (int c = 0; c < 2; ++c) {
            #pragma unroll
            for (int kt = 0; kt < 2; ++kt) {
                // B = K^T tile: lane column = key (tj + kt*16 + n); K-dim = feature
                const _Float16* kp = ws16 + KOFF +
                    ((size_t)(b * SEQ + tj + kt * 16 + n)) * DI + c * 32 + 16 * h;
                HV16 kb; kb.v = *(const v16h*)kp;
                S[kt] = __builtin_amdgcn_wmma_f32_16x16x32_f16(
                    false, Qa[c].v, false, kb.v, (short)0, S[kt], false, false);
            }
        }

        // causal mask + online softmax (row stats via 16-lane xor reduction)
        float alpha[8];
        #pragma unroll
        for (int r = 0; r < 8; ++r) {
            const int q  = t0 + r + 8 * h;
            const int k0 = tj + n, k1 = tj + 16 + n;
            float s0 = (k0 <= q) ? S[0][r] : -1.0e30f;
            float s1 = (k1 <= q) ? S[1][r] : -1.0e30f;
            float rm = fmaxf(s0, s1);
            rm = fmaxf(rm, __shfl_xor(rm, 1, 32));
            rm = fmaxf(rm, __shfl_xor(rm, 2, 32));
            rm = fmaxf(rm, __shfl_xor(rm, 4, 32));
            rm = fmaxf(rm, __shfl_xor(rm, 8, 32));
            const float mn = fmaxf(mst[r], rm);
            alpha[r] = __expf(mst[r] - mn);
            mst[r] = mn;
            const float p0 = __expf(s0 - mn);
            const float p1 = __expf(s1 - mn);
            float rs = p0 + p1;
            rs += __shfl_xor(rs, 1, 32);
            rs += __shfl_xor(rs, 2, 32);
            rs += __shfl_xor(rs, 4, 32);
            rs += __shfl_xor(rs, 8, 32);
            lsum[r] = lsum[r] * alpha[r] + rs;
            lds[(r + 8 * h) * 32 + n]      = (_Float16)p0;
            lds[(r + 8 * h) * 32 + 16 + n] = (_Float16)p1;
        }
        #pragma unroll
        for (int nt = 0; nt < 4; ++nt)
            #pragma unroll
            for (int r = 0; r < 8; ++r) O[nt][r] *= alpha[r];

        __syncthreads();
        HV16 Pa;   // P as A-fragment (transpose via LDS)
        Pa.h8[0] = *(const v8h*)&lds[m * 32 + 16 * h];
        Pa.h8[1] = *(const v8h*)&lds[m * 32 + 16 * h + 8];
        __syncthreads();

        #pragma unroll
        for (int nt = 0; nt < 4; ++nt) {
            // B = V tile: lane column = output dim, K-dim = key rows (V transposed)
            const _Float16* vp = ws16 + VOFF +
                ((size_t)(b * DI + nt * 16 + n)) * SEQ + tj + 16 * h;
            HV16 vb; vb.v = *(const v16h*)vp;
            O[nt] = __builtin_amdgcn_wmma_f32_16x16x32_f16(
                false, Pa.v, false, vb.v, (short)0, O[nt], false, false);
        }
    }

    // normalize and stage ctx (16x64) through LDS for the W0 projection
    float inv[8];
    #pragma unroll
    for (int r = 0; r < 8; ++r) inv[r] = 1.0f / lsum[r];
    #pragma unroll
    for (int nt = 0; nt < 4; ++nt)
        #pragma unroll
        for (int r = 0; r < 8; ++r)
            lds[(r + 8 * h) * 64 + nt * 16 + n] = (_Float16)(O[nt][r] * inv[r]);
    __syncthreads();

    HV16 Ca[2];
    #pragma unroll
    for (int c = 0; c < 2; ++c) {
        Ca[c].h8[0] = *(const v8h*)&lds[m * 64 + c * 32 + 16 * h];
        Ca[c].h8[1] = *(const v8h*)&lds[m * 64 + c * 32 + 16 * h + 8];
    }

    v8f acc2[4];
    #pragma unroll
    for (int nt = 0; nt < 4; ++nt) acc2[nt] = zero8();
    #pragma unroll
    for (int c = 0; c < 2; ++c) {
        #pragma unroll
        for (int nt = 0; nt < 4; ++nt) {
            HV16 wb;
            wb.v = *(const v16h*)(ws16 + W0F + ((size_t)(c * 4 + nt) * 32 + lane) * 16);
            acc2[nt] = __builtin_amdgcn_wmma_f32_16x16x32_f16(
                false, Ca[c].v, false, wb.v, (short)0, acc2[nt], false, false);
        }
    }

    #pragma unroll
    for (int nt = 0; nt < 4; ++nt) {
        const float bb = b0[nt * 16 + n];
        #pragma unroll
        for (int r = 0; r < 8; ++r)
            out[((size_t)(b * SEQ + t0 + r + 8 * h)) * DI + nt * 16 + n] =
                acc2[nt][r] + bb;
    }
}

// ---------------------------------------------------------------------------
extern "C" void kernel_launch(void* const* d_in, const int* in_sizes, int n_in,
                              void* d_out, int out_size, void* d_ws, size_t ws_size,
                              hipStream_t stream) {
    const float* x  = (const float*)d_in[0];
    const float* Wk = (const float*)d_in[1];
    const float* bk = (const float*)d_in[2];
    const float* Wq = (const float*)d_in[3];
    const float* bq = (const float*)d_in[4];
    const float* Wv = (const float*)d_in[5];
    const float* bv = (const float*)d_in[6];
    const float* W0 = (const float*)d_in[7];
    const float* b0 = (const float*)d_in[8];
    float* out = (float*)d_out;
    _Float16* ws16 = (_Float16*)d_ws;

    prep_weights<<<dim3(98, 4), 32, 0, stream>>>(Wk, Wq, Wv, W0, ws16);
    qkv_proj<<<dim3(SEQ / 16, NB), 32, 0, stream>>>(x, bk, bq, bv, ws16);
    attn_kernel<<<dim3(SEQ / 16, NB), 32, 0, stream>>>(ws16, b0, out);
}
